// BaseField_78632261256031
// MI455X (gfx1250) — matrix operand order repaired and checked by reading, not compiled, not run on hardware
//
#include <hip/hip_runtime.h>
#include <hip/hip_bf16.h>

typedef __attribute__((ext_vector_type(16))) _Float16 v16h;
typedef __attribute__((ext_vector_type(8)))  _Float16 v8h;
typedef __attribute__((ext_vector_type(8)))  float    v8f;

#define NTOT 262144
#define ROWS_WG 128
#define HSTR 264   // 256 + 8 halfs pad (keeps 16B alignment, rotates banks)
#define XSTR 72    // 64 + 8 halfs pad

// ---- f16 transposed-weight offsets in d_ws (units: halfs) ----
#define OFF0   0        // [256][64]   W0^T (63->64 pad)
#define OFF1   16384    // [256][256]
#define OFF2   81920    // [256][256]
#define OFF3   147456   // [256][256]
#define OFF4X  212992   // [256][64]   x-part of skip layer (63->64 pad)
#define OFF4H  229376   // [256][256]  h-part of skip layer
#define OFF5   294912   // [256][256]
#define OFF6   360448   // [256][256]
#define OFF7   425984   // [16][256]   (4->16 pad)
#define OFFN0  430080   // [64][256]
#define OFFN1  446464   // [64][64]
#define OFFN2  450560   // [16][64]    (3->16 pad)
#define WTOT   451584
#define BIAS_BYTE_OFF (WTOT * 2)
// ---- padded f32 bias offsets (units: floats, from BIAS_BYTE_OFF) ----
// b0..b6 at i*256
#define PB7    1792   // 16
#define PNB0   1808   // 64
#define PNB1   1872   // 64
#define PNB2   1936   // 16

// ======================= weight prep kernel =======================
struct WPtrs {
    const float* W[8];
    const float* b[8];
    const float* nW[3];
    const float* nb[3];
};

// dst[n*Kpad + k] = (k<K && n<Nsrc) ? W[k*Nsrc + n] : 0   (f16)
static __device__ __forceinline__ void tp(const float* __restrict__ W, int K, int Nsrc,
                                          _Float16* __restrict__ dst, int Nrows, int Kpad,
                                          int tid, int nth)
{
    int total = Nrows * Kpad;
    for (int i = tid; i < total; i += nth) {
        int n = i / Kpad;
        int k = i - n * Kpad;
        float v = (k < K && n < Nsrc) ? W[(size_t)k * Nsrc + n] : 0.f;
        dst[i] = (_Float16)v;
    }
}

static __device__ __forceinline__ void cp(const float* __restrict__ b, int n,
                                          float* __restrict__ dst, int npad,
                                          int tid, int nth)
{
    for (int i = tid; i < npad; i += nth) dst[i] = (i < n) ? b[i] : 0.f;
}

__global__ void prep_kernel(WPtrs p, _Float16* __restrict__ wt, float* __restrict__ pb)
{
    int tid = blockIdx.x * blockDim.x + threadIdx.x;
    int nth = gridDim.x * blockDim.x;
    tp(p.W[0],          63, 256, wt + OFF0,  256,  64, tid, nth);
    tp(p.W[1],         256, 256, wt + OFF1,  256, 256, tid, nth);
    tp(p.W[2],         256, 256, wt + OFF2,  256, 256, tid, nth);
    tp(p.W[3],         256, 256, wt + OFF3,  256, 256, tid, nth);
    tp(p.W[4],          63, 256, wt + OFF4X, 256,  64, tid, nth); // x rows 0..62
    tp(p.W[4] + 63*256,256, 256, wt + OFF4H, 256, 256, tid, nth); // h rows 63..318
    tp(p.W[5],         256, 256, wt + OFF5,  256, 256, tid, nth);
    tp(p.W[6],         256, 256, wt + OFF6,  256, 256, tid, nth);
    tp(p.W[7],         256,   4, wt + OFF7,   16, 256, tid, nth);
    tp(p.nW[0],        256,  64, wt + OFFN0,  64, 256, tid, nth);
    tp(p.nW[1],         64,  64, wt + OFFN1,  64,  64, tid, nth);
    tp(p.nW[2],         64,   3, wt + OFFN2,  16,  64, tid, nth);
    for (int i = 0; i < 7; ++i) cp(p.b[i], 256, pb + i*256, 256, tid, nth);
    cp(p.b[7],  4, pb + PB7,  16, tid, nth);
    cp(p.nb[0],64, pb + PNB0, 64, tid, nth);
    cp(p.nb[1],64, pb + PNB1, 64, tid, nth);
    cp(p.nb[2], 3, pb + PNB2, 16, tid, nth);
}

// ======================= fused MLP kernel =========================
// A fragment (16x32 f16): lanes 0-15 hold K 0..15,
// lanes 16-31 hold K 16..31  -> two 16B LDS loads per k-tile.
// B fragment (32x16 f16): lane holds 16 consecutive K for N = lane%16
// (half-wave K offset 16) -> one contiguous 32B global load from W^T.
template<int KT>
static __device__ __forceinline__ v8f acc_tiles(v8f acc,
    const _Float16* __restrict__ ain, int astride,
    const _Float16* __restrict__ wlane, int lane)
{
    const int arow = lane & 15;
    const int ah   = lane >> 4;
    const _Float16* ap = ain + arow * astride + ah * 16;
#pragma unroll
    for (int kt = 0; kt < KT; ++kt) {
        v8h alo = *(const v8h*)(ap + kt * 32);
        v8h ahi = *(const v8h*)(ap + kt * 32 + 8);
        v16h a = __builtin_shufflevector(alo, ahi, 0,1,2,3,4,5,6,7,8,9,10,11,12,13,14,15);
        v16h b = *(const v16h*)(wlane + kt * 32);
        acc = __builtin_amdgcn_wmma_f32_16x16x32_f16(false, a, false, b,
                                                     (short)0, acc, false, false);
    }
    return acc;
}

// standard hidden layer: in(LDS) x W^T(global f16) + bias -> out(LDS), optional ReLU
template<int KT, int NT, bool RELU>
static __device__ __forceinline__ void layer(
    const _Float16* __restrict__ in, int istride,
    const _Float16* __restrict__ wt, int kpad, const float* __restrict__ bias,
    _Float16* __restrict__ outp, int lane)
{
    const int nc = lane & 15;
    const int ah = lane >> 4;
    for (int nt = 0; nt < NT; ++nt) {
        int n = nt * 16 + nc;
        float bv = bias[n];
        v8f acc = {bv, bv, bv, bv, bv, bv, bv, bv};
        acc = acc_tiles<KT>(acc, in, istride, wt + (size_t)n * kpad + ah * 16, lane);
#pragma unroll
        for (int v = 0; v < 8; ++v) {
            float f = acc[v];
            if (RELU) f = fmaxf(f, 0.f);
            outp[(v + 8 * ah) * HSTR + n] = (_Float16)f;  // C layout: M=v+8*ah, N=n
        }
    }
}

__global__ __launch_bounds__(256, 1) void nerf_mlp_kernel(
    const float* __restrict__ x,
    const _Float16* __restrict__ wt,
    const float* __restrict__ pb,
    float* __restrict__ out)
{
    __shared__ __align__(16) _Float16 XS[ROWS_WG * XSTR];
    __shared__ __align__(16) _Float16 HA[ROWS_WG * HSTR];
    __shared__ __align__(16) _Float16 HB[ROWS_WG * HSTR];

    const int tid  = threadIdx.x;
    const int wid  = tid >> 5;
    const int lane = tid & 31;
    const int gbase = blockIdx.x * ROWS_WG + wid * 16;  // global row of strip start
    const int nc = lane & 15;
    const int ah = lane >> 4;

    _Float16* xsw = XS + wid * 16 * XSTR;
    _Float16* ha  = HA + wid * 16 * HSTR;
    _Float16* hb  = HB + wid * 16 * HSTR;

    // stage this wave's 16 rows of x (f32 -> f16), zero-pad col 63
    for (int i = lane; i < 16 * 64; i += 32) {
        int r = i >> 6, c = i & 63;
        float v = (c < 63) ? x[(size_t)(gbase + r) * 63 + c] : 0.f;
        xsw[r * XSTR + c] = (_Float16)v;
    }
    // strips are wave-private: no cross-wave sharing -> no barriers anywhere.

    layer<2, 16, true>(xsw, XSTR, wt + OFF0, 64,  pb + 0,    ha, lane);  // L0
    layer<8, 16, true>(ha,  HSTR, wt + OFF1, 256, pb + 256,  hb, lane);  // L1
    layer<8, 16, true>(hb,  HSTR, wt + OFF2, 256, pb + 512,  ha, lane);  // L2
    layer<8, 16, true>(ha,  HSTR, wt + OFF3, 256, pb + 768,  hb, lane);  // L3 (h3 in hb)

    // L4 (skip): [x, h3] @ W4 == x @ W4x + h3 @ W4h
    for (int nt = 0; nt < 16; ++nt) {
        int n = nt * 16 + nc;
        float bv = pb[1024 + n];
        v8f acc = {bv, bv, bv, bv, bv, bv, bv, bv};
        acc = acc_tiles<2>(acc, xsw, XSTR, wt + OFF4X + (size_t)n * 64  + ah * 16, lane);
        acc = acc_tiles<8>(acc, hb,  HSTR, wt + OFF4H + (size_t)n * 256 + ah * 16, lane);
#pragma unroll
        for (int v = 0; v < 8; ++v)
            ha[(v + 8 * ah) * HSTR + n] = (_Float16)fmaxf(acc[v], 0.f);
    }

    layer<8, 16, true>(ha, HSTR, wt + OFF5, 256, pb + 1280, hb, lane);   // L5
    layer<8, 16, true>(hb, HSTR, wt + OFF6, 256, pb + 1536, ha, lane);   // L6 (geo in ha)

    // L7: geo @ W7 -> raw[.,0..3]; density = softplus, rgb = scaled sigmoid
    {
        float bv = pb[PB7 + nc];
        v8f acc = {bv, bv, bv, bv, bv, bv, bv, bv};
        acc = acc_tiles<8>(acc, ha, HSTR, wt + OFF7 + (size_t)nc * 256 + ah * 16, lane);
#pragma unroll
        for (int v = 0; v < 8; ++v) {
            float r = acc[v];
            int row = gbase + v + 8 * ah;
            if (nc == 0) {
                // stable softplus
                out[row] = fmaxf(r, 0.f) + __logf(1.f + __expf(-fabsf(r)));
            } else if (nc < 4) {
                float sg = 1.f / (1.f + __expf(-r));
                out[(size_t)NTOT + (size_t)row * 3 + (nc - 1)] = sg * 1.002f - 0.001f;
            }
        }
    }

    // normal head: g0 = relu(geo@nW0), g1 = relu(g0@nW1), raw_n = g1@nW2
    layer<8, 4, true>(ha, HSTR, wt + OFFN0, 256, pb + PNB0, hb, lane);   // geo -> g0 (hb)
    layer<2, 4, true>(hb, HSTR, wt + OFFN1, 64,  pb + PNB1, ha, lane);   // g0 -> g1 (ha)
    {
        float bv = pb[PNB2 + nc];
        v8f acc = {bv, bv, bv, bv, bv, bv, bv, bv};
        acc = acc_tiles<2>(acc, ha, HSTR, wt + OFFN2 + (size_t)nc * 64 + ah * 16, lane);
        const int base = ah * 16;  // lanes 0..2 / 16..18 hold the 3 components
#pragma unroll
        for (int v = 0; v < 8; ++v) {
            float r = acc[v];
            // stable tanh
            float e = __expf(-2.f * fabsf(r));
            float t = copysignf((1.f - e) / (1.f + e), r);
            float s0 = __shfl(t, base + 0, 32);
            float s1 = __shfl(t, base + 1, 32);
            float s2 = __shfl(t, base + 2, 32);
            float nrm = sqrtf(s0 * s0 + s1 * s1 + s2 * s2);
            float val = t / fmaxf(nrm, 1e-12f);
            if (nc < 3) {
                int row = gbase + v + 8 * ah;
                out[(size_t)4 * NTOT + (size_t)row * 3 + nc] = val;
            }
        }
    }
}

// ============================ launch ==============================
extern "C" void kernel_launch(void* const* d_in, const int* in_sizes, int n_in,
                              void* d_out, int out_size, void* d_ws, size_t ws_size,
                              hipStream_t stream)
{
    (void)in_sizes; (void)n_in; (void)out_size; (void)ws_size;
    WPtrs p;
    const float* x = (const float*)d_in[0];
    for (int i = 0; i < 8; ++i) {
        p.W[i] = (const float*)d_in[1 + 2 * i];
        p.b[i] = (const float*)d_in[2 + 2 * i];
    }
    for (int i = 0; i < 3; ++i) {
        p.nW[i] = (const float*)d_in[17 + 2 * i];
        p.nb[i] = (const float*)d_in[18 + 2 * i];
    }
    _Float16* wt = (_Float16*)d_ws;
    float* pb = (float*)((char*)d_ws + BIAS_BYTE_OFF);

    prep_kernel<<<512, 256, 0, stream>>>(p, wt, pb);
    nerf_mlp_kernel<<<NTOT / ROWS_WG, 256, 0, stream>>>(x, wt, pb, (float*)d_out);
}
